// AdvVQVAE_27118423507214
// MI455X (gfx1250) — matrix operand order repaired and checked by reading, not compiled, not run on hardware
//
#include <hip/hip_runtime.h>
#include <hip/hip_bf16.h>

// ---------------------------------------------------------------------------
// Types for CDNA5 WMMA (wave32): v_wmma_f32_16x16x32_bf16
// ---------------------------------------------------------------------------
typedef __bf16 bf16_t;
typedef bf16_t v16bf __attribute__((ext_vector_type(16)));
typedef float  v8f   __attribute__((ext_vector_type(8)));
typedef unsigned int u32x4 __attribute__((ext_vector_type(4)));
typedef int i32x4 __attribute__((ext_vector_type(4)));
typedef int i32x8 __attribute__((ext_vector_type(8)));

#if defined(__has_builtin)
#  if __has_builtin(__builtin_amdgcn_tensor_load_to_lds)
#    define VQVAE_HAVE_TDM 1
#  endif
#  if __has_builtin(__builtin_amdgcn_s_wait_tensorcnt)
#    define VQVAE_HAVE_TENSORCNT 1
#  endif
#endif
#if __has_include(<hip/amd_detail/amd_gfx1250_TDM.h>)
#  define VQVAE_TDM_6ARG 1      // therock headers -> 6-arg builtin
#endif

// Native bf16 convert (hardware v_cvt_pk_bf16_f32 on gfx1250) + pack.
__device__ __forceinline__ unsigned int pack2bf(float lo, float hi) {
  bf16_t a = (bf16_t)lo;
  bf16_t b = (bf16_t)hi;
  unsigned short ua = __builtin_bit_cast(unsigned short, a);
  unsigned short ub = __builtin_bit_cast(unsigned short, b);
  return (unsigned int)ua | ((unsigned int)ub << 16);
}

// ---------------------------------------------------------------------------
// Implicit-GEMM conv2d (NCHW / OIHW), bf16 WMMA, fp32 accumulate, fused bias.
// Block = 256 threads = 8 waves; block owns a 16-pixel M tile (all Ho*Wo are
// multiples of 16 -> tile never crosses an image). Wave w owns couts
// [(blockIdx.y*8+w)*16, +16).
//   A: im2col 16x32 bf16 tile, double-buffered in LDS, one barrier/step.
//   B: 128x32 fp32 weight super-tile DMA'd by the Tensor Data Mover into LDS
//      (double-buffered), rows padded by TDM to stride 34 dwords (bank skew);
//      OOB rows/cols (partial Cout, 7x7 K-tail) zero-filled by TDM.
// ---------------------------------------------------------------------------
template <int KH, int KW>
__global__ __launch_bounds__(256) void vqvae_conv2d_wmma(
    const float* __restrict__ x, const float* __restrict__ w,
    const float* __restrict__ bias, float* __restrict__ out,
    int Cin, int H, int W, int Cout, int stride, int pad, int Ho, int Wo)
{
  constexpr int KHW  = KH * KW;
  constexpr bool TAIL = (KHW % 32) != 0;   // only the 7x7 stem has K % 32 != 0
  constexpr int BSTRIDE = 34;              // 32 dwords + 2 pad dwords per row
  const int HoWo = Ho * Wo;
  const int K    = Cin * KHW;
  const int lane = threadIdx.x & 31;
  const int wave = threadIdx.x >> 5;
  const int hi   = lane >> 4;
  const int mtile  = blockIdx.x;
  const int nbase0 = blockIdx.y * 128;                 // block cout base
  const int nloc   = wave * 16 + (lane & 15);          // local cout row 0..127
  const int nc     = nbase0 + nloc;

  __shared__ unsigned int a_lds[2][16 * 16];           // 2 x 1 KB
  __shared__ float        b_lds[2][128 * BSTRIDE];     // 2 x 17 KB (TDM-padded)

  // ---- K-invariant im2col coordinates for this thread's A-fill slot ----
  const int ma   = threadIdx.x >> 4;        // row 0..15 of A tile
  const int dwa  = threadIdx.x & 15;        // dword 0..15 within row
  const int pix0 = mtile * 16;              // tile base pixel
  const int img  = pix0 / HoWo;             // whole tile in one image
  const int remA = pix0 + ma - img * HoWo;
  const int oyA  = remA / Wo;
  const int oxA  = remA - oyA * Wo;
  const int iy0  = oyA * stride - pad;
  const int ix0  = oxA * stride - pad;
  const float* xb = x + (size_t)img * Cin * H * W;

  auto fillA = [&](int kkf, int bufi) {
    const int k0 = kkf + 2 * dwa;
    float v0 = 0.f, v1 = 0.f;
    #pragma unroll
    for (int e = 0; e < 2; ++e) {
      const int k = k0 + e;
      if (!TAIL || k < K) {
        const int cin = k / KHW;            // constant divisor
        const int r2  = k - cin * KHW;
        const int ky  = r2 / KW;            // constant divisor
        const int kx  = r2 - ky * KW;
        const int iy  = iy0 + ky;
        const int ix  = ix0 + kx;
        if (iy >= 0 && iy < H && ix >= 0 && ix < W) {
          const float v = xb[((size_t)cin * H + iy) * W + ix];
          if (e) v1 = v; else v0 = v;
        }
      }
    }
    a_lds[bufi][ma * 16 + dwa] = pack2bf(v0, v1);
  };

  // ---- B staging: Tensor Data Mover global->LDS (wave 0 issues) ----
  auto stageB = [&](int kkf, int bufi) {
#if defined(VQVAE_HAVE_TDM)
    if (wave == 0) {
      const float* gsrc = w + (size_t)nbase0 * K + kkf;   // tile start
      // LDS aperture: low 32 bits of a generic LDS pointer = LDS byte offset
      const unsigned lds_off =
          (unsigned)(unsigned long long)reinterpret_cast<unsigned long long>(&b_lds[bufi][0]);
      const unsigned long long ga = reinterpret_cast<unsigned long long>(gsrc);
      const unsigned td0 = (unsigned)(K - kkf);           // remaining K elems
      const unsigned td1 = (unsigned)(Cout - nbase0);     // remaining rows
      u32x4 g0;
      g0[0] = 1u;                                         // count=1, user desc
      g0[1] = lds_off;                                    // lds_addr
      g0[2] = (unsigned)ga;                               // global_addr[31:0]
      g0[3] = (unsigned)((ga >> 32) & 0x01FFFFFFull) | 0x80000000u; // [56:32]|type=2
      i32x8 g1;
      // data_size=2(4B) | pad_enable | pad_interval=4(32dw) | pad_amount=1(2dw)
      g1[0] = (int)0x03120000u;
      g1[1] = (int)(td0 << 16);                           // tensor_dim0[15:0]
      g1[2] = (int)((td0 >> 16) | (td1 << 16));           // td0[31:16] | td1[15:0]
      g1[3] = (int)((td1 >> 16) | (32u << 16));           // td1[31:16] | tile_dim0=32
      g1[4] = (int)128;                                   // tile_dim1=128, tile_dim2=0
      g1[5] = (int)(unsigned)K;                           // tensor_dim0_stride[31:0]
      g1[6] = 0;
      g1[7] = 0;
      i32x4 z4 = {0, 0, 0, 0};
#if defined(VQVAE_TDM_6ARG)
      i32x8 z8 = {0, 0, 0, 0, 0, 0, 0, 0};
      __builtin_amdgcn_tensor_load_to_lds(g0, g1, z4, z4, z8, 0);
#else
      __builtin_amdgcn_tensor_load_to_lds(g0, g1, z4, z4, 0);
#endif
    }
#else
    // cooperative fallback: 256 threads copy 128x32 tile with zero-fill
    const int rows = Cout - nbase0;
    for (int t = threadIdx.x; t < 128 * 32; t += 256) {
      const int r = t >> 5, kx = t & 31;
      const int k = kkf + kx;
      float v = 0.f;
      if (r < rows && (!TAIL || k < K)) v = w[(size_t)(nbase0 + r) * K + k];
      b_lds[bufi][r * BSTRIDE + kx] = v;
    }
#endif
  };
  auto waitB = [&]() {
#if defined(VQVAE_HAVE_TDM) && defined(VQVAE_HAVE_TENSORCNT)
    if (wave == 0) __builtin_amdgcn_s_wait_tensorcnt(0);
#endif
  };

  v8f acc = {0.f,0.f,0.f,0.f,0.f,0.f,0.f,0.f};

  // prefill both A and B buffer 0
  fillA(0, 0);
  stageB(0, 0);
  waitB();
  __syncthreads();

  int cur = 0;
  for (int kk = 0; kk < K; kk += 32) {
    // ---- A fragment from a_lds[cur]: ISA 16-bit A 16x32 layout ----
    union { v16bf v; unsigned int u[8]; } A;
    {
      const int mr = lane & 15;
      #pragma unroll
      for (int j = 0; j < 8; ++j) {
        const int kp = (hi ? 4 : 0) + (j & 3) + ((j >= 4) ? 8 : 0);
        A.u[j] = a_lds[cur][mr * 16 + kp];
      }
    }

    // ---- B fragment from b_lds[cur]: lane = cout column, K pairs ----
    union { v16bf v; unsigned int u[8]; } B;
    {
      const float* brow = &b_lds[cur][nloc * BSTRIDE + (hi ? 8 : 0)];
      #pragma unroll
      for (int j = 0; j < 8; ++j)
        B.u[j] = pack2bf(brow[2 * j + ((j >= 4) ? 8 : 0)], brow[2 * j + 1 + ((j >= 4) ? 8 : 0)]);
    }

    // ---- stage next tiles into the other buffers while these are live ----
    if (kk + 32 < K) {
      fillA(kk + 32, cur ^ 1);
      stageB(kk + 32, cur ^ 1);
      waitB();
    }
    __syncthreads();   // one barrier per step: fills done + reads of [cur] done

    acc = __builtin_amdgcn_wmma_f32_16x16x32_bf16(false, A.v, false, B.v,
                                                  (short)0, acc, false, false);
    cur ^= 1;
  }

  // ---- epilogue: out_idx = p + (img*(Cout-1) + nc) * HoWo ----
  if (nc < Cout) {
    const float bv = bias ? bias[nc] : 0.f;
    const size_t obase = ((size_t)img * (Cout - 1) + nc) * HoWo;
    #pragma unroll
    for (int j = 0; j < 8; ++j) {
      const int p = pix0 + j + (hi ? 8 : 0);
      out[obase + p] = acc[j] + bv;
    }
  }
}

// ---------------------------------------------------------------------------
// GroupNorm (+ optional fused ReLU). One block per (n, group).
// ---------------------------------------------------------------------------
__global__ __launch_bounds__(256) void vqvae_group_norm(
    const float* __restrict__ in, float* __restrict__ out,
    const float* __restrict__ gamma, const float* __restrict__ beta,
    int NB, int C, int HW2, int groups, int relu)
{
  const int n   = blockIdx.x / groups;
  const int g   = blockIdx.x - n * groups;
  const int cpg = C / groups;
  const size_t cnt = (size_t)cpg * HW2;
  const float* base = in  + ((size_t)n * C + (size_t)g * cpg) * HW2;
  float*       ob   = out + ((size_t)n * C + (size_t)g * cpg) * HW2;

  float s = 0.f, s2 = 0.f;
  for (size_t i = threadIdx.x; i < cnt; i += 256) {
    const float v = base[i]; s += v; s2 += v * v;
  }
  __shared__ float sh[256], sh2[256];
  sh[threadIdx.x] = s; sh2[threadIdx.x] = s2;
  __syncthreads();
  for (int o = 128; o > 0; o >>= 1) {
    if ((int)threadIdx.x < o) {
      sh[threadIdx.x]  += sh[threadIdx.x + o];
      sh2[threadIdx.x] += sh2[threadIdx.x + o];
    }
    __syncthreads();
  }
  const float mean = sh[0] / (float)cnt;
  const float var  = sh2[0] / (float)cnt - mean * mean;
  const float inv  = rsqrtf(var + 1e-5f);
  for (size_t i = threadIdx.x; i < cnt; i += 256) {
    const int c = g * cpg + (int)(i / HW2);
    float v = (base[i] - mean) * inv * gamma[c] + beta[c];
    if (relu) v = fmaxf(v, 0.f);
    ob[i] = v;
  }
}

__global__ void vqvae_add_relu(const float* __restrict__ a, const float* __restrict__ b,
                               float* __restrict__ out, size_t n)
{
  for (size_t i = (size_t)blockIdx.x * blockDim.x + threadIdx.x; i < n;
       i += (size_t)gridDim.x * blockDim.x)
    out[i] = fmaxf(a[i] + b[i], 0.f);
}

// Bilinear 2x upsample, half-pixel centers (align_corners=False).
__global__ void vqvae_upsample2x(const float* __restrict__ in, float* __restrict__ out,
                                 int NB, int C, int H, int W)
{
  const int Ho = 2 * H, Wo = 2 * W;
  const size_t total = (size_t)NB * C * Ho * Wo;
  for (size_t i = (size_t)blockIdx.x * blockDim.x + threadIdx.x; i < total;
       i += (size_t)gridDim.x * blockDim.x) {
    const int ox = (int)(i % Wo); size_t t = i / Wo;
    const int oy = (int)(t % Ho); t /= Ho;
    const int c  = (int)(t % C);
    const int n  = (int)(t / C);
    const float fy = (oy + 0.5f) * 0.5f - 0.5f;
    const float fx = (ox + 0.5f) * 0.5f - 0.5f;
    int y0 = (int)floorf(fy), x0 = (int)floorf(fx);
    const float wy = fy - (float)y0, wx = fx - (float)x0;
    int y1 = y0 + 1, x1 = x0 + 1;
    y0 = max(0, min(H - 1, y0)); y1 = max(0, min(H - 1, y1));
    x0 = max(0, min(W - 1, x0)); x1 = max(0, min(W - 1, x1));
    const float* b = in + ((size_t)n * C + c) * H * W;
    const float v = (1.f - wy) * ((1.f - wx) * b[y0 * W + x0] + wx * b[y0 * W + x1])
                  +        wy  * ((1.f - wx) * b[y1 * W + x0] + wx * b[y1 * W + x1]);
    out[i] = v;
  }
}

__global__ void vqvae_emb_norm(const float* __restrict__ emb, float* __restrict__ en,
                               int ncode, int C)
{
  const int n = blockIdx.x * blockDim.x + threadIdx.x;
  if (n < ncode) {
    const float* r = emb + (size_t)n * C;
    float s = 0.f;
    for (int c = 0; c < C; ++c) { const float v = r[c]; s += v * v; }
    en[n] = s;
  }
}

// ---------------------------------------------------------------------------
// VQ argmin: score(m,n) = ||e_n||^2 - 2 * <z_m, e_n>  via WMMA GEMM.
// Block = 8 waves owns a 16-position tile; wave w covers code tiles
// {w, w+8, w+16, w+24} -> all 512 codes. Scores in LDS, first-min argmin.
// ---------------------------------------------------------------------------
__global__ __launch_bounds__(256) void vqvae_vq_argmin(
    const float* __restrict__ z, const float* __restrict__ emb,
    const float* __restrict__ en, int* __restrict__ idx,
    int C, int HW2, int NCODE)
{
  const int lane = threadIdx.x & 31;
  const int wave = threadIdx.x >> 5;
  const int hi   = lane >> 4;
  const int mtile = blockIdx.x;

  __shared__ unsigned int a_lds[2][16 * 16];
  __shared__ float scores[16 * 512];

  // K-invariant A-fill coordinates
  const int ma  = threadIdx.x >> 4;
  const int dwa = threadIdx.x & 15;
  const int p   = mtile * 16 + ma;
  const int img = p / HW2;                  // HW2 multiple of 16
  const int rem = p - img * HW2;
  const float* zb = z + (size_t)img * C * HW2 + rem;

  auto fillA = [&](int kkf, int bufi) {
    const int k0 = kkf + 2 * dwa;
    const float v0 = zb[(size_t)k0 * HW2];
    const float v1 = zb[(size_t)(k0 + 1) * HW2];
    a_lds[bufi][ma * 16 + dwa] = pack2bf(v0, v1);
  };

  const v8f zero = {0.f,0.f,0.f,0.f,0.f,0.f,0.f,0.f};
  v8f acc[4] = {zero, zero, zero, zero};

  fillA(0, 0);
  __syncthreads();

  int cur = 0;
  for (int kk = 0; kk < C; kk += 32) {
    union { v16bf v; unsigned int u[8]; } A;
    const int mr = lane & 15;
    #pragma unroll
    for (int j = 0; j < 8; ++j)
      A.u[j] = a_lds[cur][mr * 16 + (hi ? 4 : 0) + (j & 3) + ((j >= 4) ? 8 : 0)];

    union { v16bf v; unsigned int u[8]; } Bf[4];
    #pragma unroll
    for (int s = 0; s < 4; ++s) {
      const int n = (wave + s * 8) * 16 + (lane & 15);
      const float2* er = reinterpret_cast<const float2*>(emb + (size_t)n * C + kk + (hi ? 8 : 0));
      #pragma unroll
      for (int j = 0; j < 8; ++j) {
        const float2 f = er[j + ((j >= 4) ? 4 : 0)];
        Bf[s].u[j] = pack2bf(f.x, f.y);
      }
    }

    if (kk + 32 < C) fillA(kk + 32, cur ^ 1);
    __syncthreads();

    #pragma unroll
    for (int s = 0; s < 4; ++s)
      acc[s] = __builtin_amdgcn_wmma_f32_16x16x32_bf16(false, A.v, false, Bf[s].v,
                                                       (short)0, acc[s], false, false);
    cur ^= 1;
  }

  #pragma unroll
  for (int s = 0; s < 4; ++s) {
    const int n = (wave + s * 8) * 16 + (lane & 15);
    const float enn = en[n];
    #pragma unroll
    for (int j = 0; j < 8; ++j) {
      const int m = j + (hi ? 8 : 0);
      scores[m * 512 + n] = enn - 2.f * acc[s][j];
    }
  }
  __syncthreads();

  if (threadIdx.x < 16) {
    const int m = threadIdx.x;
    float best = scores[m * 512]; int bi = 0;
    for (int n = 1; n < NCODE; ++n) {
      const float v = scores[m * 512 + n];
      if (v < best) { best = v; bi = n; }   // first-min, matches jnp.argmin
    }
    idx[mtile * 16 + m] = bi;
  }
}

__global__ void vqvae_zero1(float* p) { if (threadIdx.x == 0 && blockIdx.x == 0) p[0] = 0.f; }

// Gather q = emb[idx] into NCHW + reduce sum((z-q)^2) -> accum.
__global__ __launch_bounds__(256) void vqvae_vq_gather_loss(
    const float* __restrict__ z, const float* __restrict__ emb,
    const int* __restrict__ idx, float* __restrict__ q,
    float* __restrict__ accum, int P, int C, int HW2)
{
  const size_t total = (size_t)P * C;
  float local = 0.f;
  for (size_t i = (size_t)blockIdx.x * blockDim.x + threadIdx.x; i < total;
       i += (size_t)gridDim.x * blockDim.x) {
    const int c    = (int)(i % C);
    const size_t p = i / C;
    const int img  = (int)(p / HW2);
    const int rem  = (int)(p - (size_t)img * HW2);
    const float e  = emb[(size_t)idx[p] * C + c];
    const size_t zi = ((size_t)img * C + c) * HW2 + rem;
    const float d = z[zi] - e;
    q[zi] = e;
    local += d * d;
  }
  __shared__ float sh[256];
  sh[threadIdx.x] = local;
  __syncthreads();
  for (int o = 128; o > 0; o >>= 1) {
    if ((int)threadIdx.x < o) sh[threadIdx.x] += sh[threadIdx.x + o];
    __syncthreads();
  }
  if (threadIdx.x == 0) atomicAdd(accum, sh[0]);
}

__global__ void vqvae_finalize_losses(const float* __restrict__ acc,
                                      float* __restrict__ out, float inv)
{
  if (threadIdx.x == 0 && blockIdx.x == 0) {
    const float m = acc[0] * inv;
    out[0] = 0.25f * m;   // commit_loss  (BETA * mean)
    out[1] = m;           // codebook_loss (same mean in fwd)
  }
}

__global__ void vqvae_tanh(const float* __restrict__ in, float* __restrict__ out, size_t n)
{
  for (size_t i = (size_t)blockIdx.x * blockDim.x + threadIdx.x; i < n;
       i += (size_t)gridDim.x * blockDim.x)
    out[i] = tanhf(in[i]);
}

// ---------------------------------------------------------------------------
// Host-side orchestration
// ---------------------------------------------------------------------------
enum {
  IN_X = 0,
  IN_STEM_W, IN_STEM_B, IN_GN1_G, IN_GN1_B,
  IN_ERES1,                                    // 10: w1,b1,g1,bt1,w2,b2,g2,bt2,ws,bs
  IN_EDOWN1_W = IN_ERES1 + 10, IN_EDOWN1_B,
  IN_ERES2,                                    // 10
  IN_EDOWN2_W = IN_ERES2 + 10, IN_EDOWN2_B,
  IN_ERES3,                                    // 10
  IN_ERES4 = IN_ERES3 + 10,                    // 8 (256->256, no shortcut)
  IN_EMB = IN_ERES4 + 8,
  IN_DRES1,                                    // 8 (256->256)
  IN_DRES2 = IN_DRES1 + 8,                     // 10
  IN_DCONV2_W = IN_DRES2 + 10, IN_DCONV2_B, IN_DGN2_G, IN_DGN2_B,
  IN_DRES3,                                    // 10
  IN_DCONV3_W = IN_DRES3 + 10, IN_DCONV3_B, IN_DGN3_G, IN_DGN3_B,
  IN_DRES4,                                    // 10
  IN_DCONV4_W = IN_DRES4 + 10, IN_DCONV4_B
};

struct RBP { const float *w1,*b1,*g1,*bt1,*w2,*b2,*g2,*bt2,*ws,*bs; };

static inline RBP get_rbp(void* const* din, int base, bool sc) {
  RBP r;
  r.w1  = (const float*)din[base + 0]; r.b1  = (const float*)din[base + 1];
  r.g1  = (const float*)din[base + 2]; r.bt1 = (const float*)din[base + 3];
  r.w2  = (const float*)din[base + 4]; r.b2  = (const float*)din[base + 5];
  r.g2  = (const float*)din[base + 6]; r.bt2 = (const float*)din[base + 7];
  r.ws  = sc ? (const float*)din[base + 8] : nullptr;
  r.bs  = sc ? (const float*)din[base + 9] : nullptr;
  return r;
}

static inline void launch_conv(hipStream_t s, const float* x, const float* w, const float* b,
                               float* out, int NB, int Cin, int H, int W, int Cout,
                               int KH, int KW, int stride, int pad)
{
  const int Ho = (H + 2 * pad - KH) / stride + 1;
  const int Wo = (W + 2 * pad - KW) / stride + 1;
  const int Mtot = NB * Ho * Wo;                 // Ho*Wo % 16 == 0 for all layers
  dim3 grid(Mtot / 16, (Cout + 127) / 128);
  if (KH == 7)
    vqvae_conv2d_wmma<7, 7><<<grid, 256, 0, s>>>(x, w, b, out, Cin, H, W, Cout, stride, pad, Ho, Wo);
  else if (KH == 3)
    vqvae_conv2d_wmma<3, 3><<<grid, 256, 0, s>>>(x, w, b, out, Cin, H, W, Cout, stride, pad, Ho, Wo);
  else
    vqvae_conv2d_wmma<1, 1><<<grid, 256, 0, s>>>(x, w, b, out, Cin, H, W, Cout, stride, pad, Ho, Wo);
}

static inline void launch_gn(hipStream_t s, const float* in, float* out,
                             const float* g, const float* b,
                             int NB, int C, int H, int W, int relu)
{
  vqvae_group_norm<<<NB * 32, 256, 0, s>>>(in, out, g, b, NB, C, H * W, 32, relu);
}

static inline void launch_add_relu(hipStream_t s, const float* a, const float* b,
                                   float* out, size_t n)
{
  const int blocks = (int)((n + 255) / 256);
  vqvae_add_relu<<<blocks, 256, 0, s>>>(a, b, out, n);
}

static inline void launch_upsample(hipStream_t s, const float* in, float* out,
                                   int NB, int C, int H, int W)
{
  const size_t total = (size_t)NB * C * 4 * H * W;
  const int blocks = (int)((total + 255) / 256);
  vqvae_upsample2x<<<blocks, 256, 0, s>>>(in, out, NB, C, H, W);
}

static inline void run_res_block(hipStream_t s, const RBP& p, const float* in, float* out,
                                 float* t1, float* t2, float* tsc,
                                 int NB, int Cin, int H, int W, int Cout)
{
  const float* idn = in;
  if (p.ws) { launch_conv(s, in, p.ws, p.bs, tsc, NB, Cin, H, W, Cout, 1, 1, 1, 0); idn = tsc; }
  launch_conv(s, in, p.w1, p.b1, t1, NB, Cin, H, W, Cout, 3, 3, 1, 1);
  launch_gn(s, t1, t1, p.g1, p.bt1, NB, Cout, H, W, 1);
  launch_conv(s, t1, p.w2, p.b2, t2, NB, Cout, H, W, Cout, 3, 3, 1, 1);
  launch_gn(s, t2, t2, p.g2, p.bt2, NB, Cout, H, W, 0);
  launch_add_relu(s, t2, idn, out, (size_t)NB * Cout * H * W);
}

extern "C" void kernel_launch(void* const* d_in, const int* in_sizes, int n_in,
                              void* d_out, int out_size, void* d_ws, size_t ws_size,
                              hipStream_t stream)
{
  (void)in_sizes; (void)n_in; (void)out_size; (void)ws_size;
  const int NB = 16;
  auto F = [&](int i) { return (const float*)d_in[i]; };

  // ---- workspace carving (~405 MB) ----
  char* ws = (char*)d_ws;
  const size_t T_BYTES = 16ull * 64 * 128 * 128 * 4;   // 67,108,864 B per temp
  float* T0  = (float*)(ws);
  float* T1  = (float*)(ws + T_BYTES * 1);
  float* T2  = (float*)(ws + T_BYTES * 2);
  float* T3  = (float*)(ws + T_BYTES * 3);
  float* BIG = (float*)(ws + T_BYTES * 4);             // 16*32*256*256 floats
  char*  tail = ws + T_BYTES * 4 + 16ull * 32 * 256 * 256 * 4;
  float* EN  = (float*)tail;                           // 512 fp32
  int*   IDX = (int*)(tail + 4096);                    // 16384 int32
  float* ACC = (float*)(tail + 4096 + 65536);          // 1 fp32

  // ================= Encoder =================
  // stem: 3->32, 7x7 s2 p3, 256 -> 128 ; GN+ReLU
  launch_conv(stream, F(IN_X), F(IN_STEM_W), F(IN_STEM_B), T0, NB, 3, 256, 256, 32, 7, 7, 2, 3);
  launch_gn(stream, T0, T0, F(IN_GN1_G), F(IN_GN1_B), NB, 32, 128, 128, 1);
  // e_res1: 32->64 @128
  run_res_block(stream, get_rbp(d_in, IN_ERES1, true), T0, T0, T2, T3, T1, NB, 32, 128, 128, 64);
  // e_down1: 64->64, 3x3 s2, 128 -> 64
  launch_conv(stream, T0, F(IN_EDOWN1_W), F(IN_EDOWN1_B), T1, NB, 64, 128, 128, 64, 3, 3, 2, 1);
  // e_res2: 64->128 @64
  run_res_block(stream, get_rbp(d_in, IN_ERES2, true), T1, T1, T3, T0, T2, NB, 64, 64, 64, 128);
  // e_down2: 128->128, 3x3 s2, 64 -> 32
  launch_conv(stream, T1, F(IN_EDOWN2_W), F(IN_EDOWN2_B), T2, NB, 128, 64, 64, 128, 3, 3, 2, 1);
  // e_res3: 128->256 @32
  run_res_block(stream, get_rbp(d_in, IN_ERES3, true), T2, T2, T0, T1, T3, NB, 128, 32, 32, 256);
  // e_res4: 256->256 @32 (no shortcut conv)  -> z in T3
  run_res_block(stream, get_rbp(d_in, IN_ERES4, false), T2, T3, T0, T1, nullptr, NB, 256, 32, 32, 256);

  // ================= Vector Quantizer =================
  vqvae_emb_norm<<<2, 256, 0, stream>>>(F(IN_EMB), EN, 512, 256);
  vqvae_vq_argmin<<<16384 / 16, 256, 0, stream>>>(T3, F(IN_EMB), EN, IDX, 256, 1024, 512);
  vqvae_zero1<<<1, 32, 0, stream>>>(ACC);
  vqvae_vq_gather_loss<<<4096, 256, 0, stream>>>(T3, F(IN_EMB), IDX, T0, ACC, 16384, 256, 1024);
  vqvae_finalize_losses<<<1, 32, 0, stream>>>(ACC, (float*)d_out + 16ull * 3 * 256 * 256,
                                              1.f / (16384.f * 256.f));

  // ================= Decoder ================= (q in T0)
  // d_res1: 256->256 @32 -> T3
  run_res_block(stream, get_rbp(d_in, IN_DRES1, false), T0, T3, T1, T2, nullptr, NB, 256, 32, 32, 256);
  // d_res2: 256->128 @32 -> T1 (shortcut buffer aliases out: written first, read last)
  run_res_block(stream, get_rbp(d_in, IN_DRES2, true), T3, T1, T0, T2, T1, NB, 256, 32, 32, 128);
  // upsample 32 -> 64
  launch_upsample(stream, T1, T2, NB, 128, 32, 32);
  // d_conv2 + GN + ReLU
  launch_conv(stream, T2, F(IN_DCONV2_W), F(IN_DCONV2_B), T0, NB, 128, 64, 64, 128, 3, 3, 1, 1);
  launch_gn(stream, T0, T0, F(IN_DGN2_G), F(IN_DGN2_B), NB, 128, 64, 64, 1);
  // d_res3: 128->64 @64 -> T0
  run_res_block(stream, get_rbp(d_in, IN_DRES3, true), T0, T0, T2, T3, T1, NB, 128, 64, 64, 64);
  // upsample 64 -> 128
  launch_upsample(stream, T0, T1, NB, 64, 64, 64);
  // d_conv3 + GN + ReLU
  launch_conv(stream, T1, F(IN_DCONV3_W), F(IN_DCONV3_B), T2, NB, 64, 128, 128, 64, 3, 3, 1, 1);
  launch_gn(stream, T2, T2, F(IN_DGN3_G), F(IN_DGN3_B), NB, 64, 128, 128, 1);
  // d_res4: 64->32 @128 -> T1 (shortcut aliases out)
  run_res_block(stream, get_rbp(d_in, IN_DRES4, true), T2, T1, T0, T3, T1, NB, 64, 128, 128, 32);
  // upsample 128 -> 256
  launch_upsample(stream, T1, BIG, NB, 32, 128, 128);
  // d_conv4: 32->3 @256
  launch_conv(stream, BIG, F(IN_DCONV4_W), F(IN_DCONV4_B), T0, NB, 32, 256, 256, 3, 3, 3, 1, 1);
  // tanh -> d_out (recon)
  {
    const size_t n = 16ull * 3 * 256 * 256;
    vqvae_tanh<<<(int)((n + 255) / 256), 256, 0, stream>>>(T0, (float*)d_out, n);
  }
}
